// LSTM_without_torch_29532195127304
// MI455X (gfx1250) — matrix operand order repaired and checked by reading, not compiled, not run on hardware
//
#include <hip/hip_runtime.h>

typedef __attribute__((ext_vector_type(16))) __bf16 v16bf;
typedef __attribute__((ext_vector_type(8)))  float  v8f;

#define S_LEN   256
#define BATCH   64
#define IN_DIM  128
#define HID     256
#define ZK      384            // IN + HID
#define ZSTRIDE 392            // padded LDS row stride (elements) to break bank conflicts
#define VOCAB   10000
#define AT_ELEMS (256ULL * 64ULL * 10000ULL)

// fast inline nonlinearities: v_exp_f32 + raw v_rcp_f32 (no IEEE div expansion)
__device__ __forceinline__ float fast_rcp(float x) { return __builtin_amdgcn_rcpf(x); }
__device__ __forceinline__ float sigmoidf(float x) { return fast_rcp(1.0f + __expf(-x)); }
__device__ __forceinline__ float tanh_fast(float x) {
  return 1.0f - 2.0f * fast_rcp(__expf(2.0f * x) + 1.0f);
}

// ---------------- prep kernels: bf16 conversion / transposition / gather ----------------

// Xbf[s][b][c] = bf16(emb[input[b][s]][c]);  N = S*B*IN = 2,097,152 (divisible by 256)
__global__ void embed_gather(const int* __restrict__ tok, const float* __restrict__ emb,
                             __bf16* __restrict__ Xbf) {
  int idx = blockIdx.x * blockDim.x + threadIdx.x;
  int c = idx & (IN_DIM - 1);
  int b = (idx >> 7) & (BATCH - 1);
  int s = idx >> 13;
  int t = tok[b * S_LEN + s];
  Xbf[idx] = (__bf16)emb[(size_t)t * IN_DIM + c];
}

// WgT[g][n][k] = bf16(Wg[k][n]);  N = 4*HID*ZK = 393,216 (divisible by 256)
__global__ void prep_gate_weights(const float* __restrict__ Wf, const float* __restrict__ Wi,
                                  const float* __restrict__ Wc, const float* __restrict__ Wo,
                                  __bf16* __restrict__ WgT) {
  int idx = blockIdx.x * blockDim.x + threadIdx.x;
  int k = idx % ZK;
  int n = (idx / ZK) % HID;
  int g = idx / (ZK * HID);
  const float* Wsrc = (g == 0) ? Wf : (g == 1) ? Wi : (g == 2) ? Wc : Wo;
  WgT[idx] = (__bf16)Wsrc[(size_t)k * HID + n];
}

// WT[v][h] = bf16(W[h][v]);  N = VOCAB*HID = 2,560,000 (divisible by 256)
__global__ void prep_vocab_weights(const float* __restrict__ W, __bf16* __restrict__ WT) {
  int idx = blockIdx.x * blockDim.x + threadIdx.x;
  int h = idx & (HID - 1);
  int v = idx >> 8;
  WT[idx] = (__bf16)W[(size_t)h * VOCAB + v];
}

// ---------------- persistent recurrence kernel: 4 blocks x 16 batch rows ----------------
// Each of 16 waves owns one 16-wide HID column tile for ALL four gates; cell state c
// stays in 8 accumulator VGPRs for the whole sequence. z = [x_t | h] lives in LDS (bf16).
// Gate weights stream from L2 every K-chunk (non-unrolled loop -> no hoist/spill),
// with near-scope global_prefetch two chunks ahead.
__global__ __launch_bounds__(512) void lstm_recurrence(
    const __bf16* __restrict__ Xbf, const __bf16* __restrict__ WgT,
    const float* __restrict__ bfv, const float* __restrict__ biv,
    const float* __restrict__ bcv, const float* __restrict__ bov,
    __bf16* __restrict__ Hbf, float* __restrict__ out) {
  __shared__ __bf16 z[16 * ZSTRIDE];

  const int tid  = threadIdx.x;
  const int wave = tid >> 5;
  const int lane = tid & 31;
  const int lr   = lane & 15;   // row within tile (A) / col within tile (B,C)
  const int hi   = lane >> 4;   // lane-half selector
  const int col0 = wave * 16;   // this wave's HID column tile
  const int n    = col0 + lr;   // this lane's HID column
  const int gr0  = blockIdx.x * 16;  // global batch-row base

  // zero the h region of z (cols 128..383)
  {
    int r  = tid >> 5;
    int cc = IN_DIM + (tid & 31) * 8;
    uint4 zz; zz.x = 0; zz.y = 0; zz.z = 0; zz.w = 0;
    *(uint4*)&z[r * ZSTRIDE + cc] = zz;
  }

  const v8f zero8 = {0, 0, 0, 0, 0, 0, 0, 0};
  v8f c_acc = zero8;
  v8f hreg  = zero8;

  const float bF = bfv[n], bI = biv[n], bC = bcv[n], bO = bov[n];
  // per-lane B-fragment base addresses (column n, lane-half K offset)
  const __bf16* wgF0 = WgT + (size_t)n * ZK + hi * 16;
  const __bf16* wgI0 = wgF0 + (size_t)1 * HID * ZK;
  const __bf16* wgG0 = wgF0 + (size_t)2 * HID * ZK;
  const __bf16* wgO0 = wgF0 + (size_t)3 * HID * ZK;

  for (int s = 0; s < S_LEN; ++s) {
    // cooperative load of x_t tile (16 x 128 bf16) into z cols 0..127
    {
      int r  = tid >> 5;
      int cc = (tid & 31) * 4;
      *(uint2*)&z[r * ZSTRIDE + cc] =
          *(const uint2*)(Xbf + ((size_t)s * BATCH + gr0 + r) * IN_DIM + cc);
    }
    __syncthreads();

    v8f accF = zero8, accI = zero8, accG = zero8, accO = zero8;
    const __bf16* apz = &z[lr * ZSTRIDE + hi * 16];
    const __bf16* pF = wgF0;
    const __bf16* pI = wgI0;
    const __bf16* pG = wgG0;
    const __bf16* pO = wgO0;
#pragma unroll 1
    for (int k0 = 0; k0 < ZK / 32; ++k0) {
      // near-scope prefetch two K-chunks ahead of all four gate weight streams
      __builtin_prefetch(pF + 64, 0, 3);
      __builtin_prefetch(pI + 64, 0, 3);
      __builtin_prefetch(pG + 64, 0, 3);
      __builtin_prefetch(pO + 64, 0, 3);
      // A fragment (16x32 bf16) from LDS: two 16B chunks per lane
      v16bf a;
      ((uint4*)&a)[0] = *(const uint4*)(apz);
      ((uint4*)&a)[1] = *(const uint4*)(apz + 8);
      // B fragments (32x16 bf16): one contiguous 32B chunk per lane, from L2
      v16bf b0 = *(const v16bf*)pF;
      v16bf b1 = *(const v16bf*)pI;
      v16bf b2 = *(const v16bf*)pG;
      v16bf b3 = *(const v16bf*)pO;
      accF = __builtin_amdgcn_wmma_f32_16x16x32_bf16(false, a, false, b0, (short)0, accF, false, false);
      accI = __builtin_amdgcn_wmma_f32_16x16x32_bf16(false, a, false, b1, (short)0, accI, false, false);
      accG = __builtin_amdgcn_wmma_f32_16x16x32_bf16(false, a, false, b2, (short)0, accG, false, false);
      accO = __builtin_amdgcn_wmma_f32_16x16x32_bf16(false, a, false, b3, (short)0, accO, false, false);
      apz += 32; pF += 32; pI += 32; pG += 32; pO += 32;
    }

    // pointwise LSTM cell update (C/D layout: VGPR j, lane -> row j + 8*hi, col n)
#pragma unroll
    for (int j = 0; j < 8; ++j) {
      float f  = sigmoidf(accF[j] + bF);
      float i  = sigmoidf(accI[j] + bI);
      float g  = tanh_fast(accG[j] + bC);
      float o  = sigmoidf(accO[j] + bO);
      float cn = f * c_acc[j] + i * g;
      c_acc[j] = cn;
      hreg[j]  = o * tanh_fast(cn);
    }
    __syncthreads();  // all waves finished reading old z

    // write new h tile into z cols 128..383 (bf16)
#pragma unroll
    for (int j = 0; j < 8; ++j)
      z[(j + hi * 8) * ZSTRIDE + IN_DIM + n] = (__bf16)hreg[j];
    __syncthreads();

    // cooperative coalesced copy of h (16 x 256 bf16) to global H for the logits GEMM
    {
      int r  = tid >> 5;
      int cc = (tid & 31) * 8;
      *(uint4*)(Hbf + ((size_t)s * BATCH + gr0 + r) * HID + cc) =
          *(const uint4*)&z[r * ZSTRIDE + IN_DIM + cc];
    }
  }

  // final (hT, cT) in f32 straight from registers
#pragma unroll
  for (int j = 0; j < 8; ++j) {
    int row = gr0 + j + hi * 8;
    out[AT_ELEMS + (size_t)row * HID + n] = hreg[j];
    out[AT_ELEMS + (size_t)BATCH * HID + (size_t)row * HID + n] = c_acc[j];
  }
}

// ---------------- logits GEMM: [16384 x 256] x [256 x 10000] + bias ----------------
// grid(16, 1024), 256 threads. Wave = one 16-row M-tile x 5 N-tiles (A reused 5x);
// 625 N-tiles = 125 wave-groups x 5. HBM-bound only on the 640MB f32 output.
__global__ __launch_bounds__(256) void logits_gemm(
    const __bf16* __restrict__ Hbf, const __bf16* __restrict__ WT,
    const float* __restrict__ bias, float* __restrict__ at) {
  const int tid  = threadIdx.x;
  const int wave = tid >> 5;
  const int lane = tid & 31;
  const int lr   = lane & 15;
  const int hi   = lane >> 4;
  const int ngrp = blockIdx.x * 8 + wave;   // 0..127, each covers 5 N-tiles
  if (ngrp >= 125) return;                  // wave-uniform: EXEC stays all-ones
  const int m0 = blockIdx.y * 16;

  const v8f zero8 = {0, 0, 0, 0, 0, 0, 0, 0};
  v8f acc[5] = {zero8, zero8, zero8, zero8, zero8};

  const __bf16* arow = Hbf + ((size_t)m0 + lr) * HID;
#pragma unroll
  for (int k0 = 0; k0 < HID / 32; ++k0) {
    v16bf a;
    const __bf16* ap = arow + k0 * 32 + hi * 16;
    ((uint4*)&a)[0] = *(const uint4*)ap;
    ((uint4*)&a)[1] = *(const uint4*)(ap + 8);
#pragma unroll
    for (int t = 0; t < 5; ++t) {
      int nn = (ngrp * 5 + t) * 16 + lr;
      v16bf b = *(const v16bf*)(WT + (size_t)nn * HID + k0 * 32 + hi * 16);
      acc[t] = __builtin_amdgcn_wmma_f32_16x16x32_bf16(false, a, false, b, (short)0, acc[t], false, false);
    }
  }

#pragma unroll
  for (int t = 0; t < 5; ++t) {
    int colbase = (ngrp * 5 + t) * 16;
    float bv = bias[colbase + lr];
#pragma unroll
    for (int j = 0; j < 8; ++j) {
      int row = m0 + j + hi * 8;
      at[(size_t)row * VOCAB + colbase + lr] = acc[t][j] + bv;
    }
  }
}

// ---------------- launcher ----------------
extern "C" void kernel_launch(void* const* d_in, const int* in_sizes, int n_in,
                              void* d_out, int out_size, void* d_ws, size_t ws_size,
                              hipStream_t stream) {
  const int*   tok = (const int*)d_in[0];
  const float* emb = (const float*)d_in[1];
  const float* Wf  = (const float*)d_in[2];
  const float* Wi  = (const float*)d_in[3];
  const float* Wc  = (const float*)d_in[4];
  const float* Wo  = (const float*)d_in[5];
  const float* bf_ = (const float*)d_in[6];
  const float* bi_ = (const float*)d_in[7];
  const float* bc_ = (const float*)d_in[8];
  const float* bo_ = (const float*)d_in[9];
  const float* W   = (const float*)d_in[10];
  const float* b   = (const float*)d_in[11];
  float* out = (float*)d_out;

  // workspace layout (bytes, 256-aligned):
  //   Xbf  [S][B][IN]  bf16 : 4,194,304
  //   WgT  [4][HID][ZK] bf16:   786,432
  //   WT   [VOCAB][HID] bf16: 5,120,000
  //   Hbf  [S][B][HID] bf16 : 8,388,608
  char* ws = (char*)d_ws;
  __bf16* Xbf = (__bf16*)(ws);
  __bf16* WgT = (__bf16*)(ws + 4194304);
  __bf16* WT  = (__bf16*)(ws + 4194304 + 786432);
  __bf16* Hbf = (__bf16*)(ws + 4194304 + 786432 + 5120000);

  embed_gather<<<(S_LEN * BATCH * IN_DIM) / 256, 256, 0, stream>>>(tok, emb, Xbf);
  prep_gate_weights<<<(4 * HID * ZK) / 256, 256, 0, stream>>>(Wf, Wi, Wc, Wo, WgT);
  prep_vocab_weights<<<(VOCAB * HID) / 256, 256, 0, stream>>>(W, WT);

  lstm_recurrence<<<BATCH / 16, 512, 0, stream>>>(Xbf, WgT, bf_, bi_, bc_, bo_, Hbf, out);

  logits_gemm<<<dim3(16, (S_LEN * BATCH) / 16), 256, 0, stream>>>(Hbf, WT, b, out);
}
